// BlockDecay_12549894439573
// MI455X (gfx1250) — matrix operand order repaired and checked by reading, not compiled, not run on hardware
//
#include <hip/hip_runtime.h>
#include <hip/hip_bf16.h>

#define Bx   8
#define Wx   4096
#define Rx   128
#define Dx   128
#define BLKx 128
#define NBx  32

typedef __bf16 v16bf __attribute__((ext_vector_type(16)));
typedef float  v8f   __attribute__((ext_vector_type(8)));

union Frag {
    v16bf v;
    unsigned int u[8];
};

__device__ __forceinline__ unsigned int pack_bf16(float a, float b) {
    union { __bf16 h[2]; unsigned int u; } p;
    p.h[0] = (__bf16)a;
    p.h[1] = (__bf16)b;
    return p.u;
}

// ---- fragment loaders -------------------------------------------------------
// A-matrix 16x32 bf16 (ISA 7.12.2): lane -> M, dword v -> K pair.
//   v<4 : K = kBase + 2v + 8*half ;  v>=4 : K = kBase + 16 + 2(v-4) + 8*half
// B-matrix 32x16 bf16: lane -> N, dword v -> K pair: same K mapping as A
__device__ __forceinline__ v16bf loadA_bf16(const __bf16* mat, int strideK,
                                            int mBase, int kBase, int lane) {
    const int half = lane >> 4;
    const int m = mBase + (lane & 15);
    const unsigned int* rowp = (const unsigned int*)(mat + (size_t)m * strideK);
    Frag f;
#pragma unroll
    for (int v = 0; v < 8; ++v) {
        int k = kBase + 2 * v + 8 * half + (v >= 4 ? 8 : 0);
        f.u[v] = rowp[k >> 1];
    }
    return f.v;
}

__device__ __forceinline__ v16bf loadB_bf16(const __bf16* mat, int strideK,
                                            int nBase, int kBase, int lane) {
    const int half = lane >> 4;
    const int n = nBase + (lane & 15);
    const unsigned int* rowp = (const unsigned int*)(mat + (size_t)n * strideK);
    Frag f;
#pragma unroll
    for (int v = 0; v < 8; ++v) {
        int k = kBase + 2 * v + 8 * half + (v >= 4 ? 8 : 0);
        f.u[v] = rowp[k >> 1];
    }
    return f.v;
}

__device__ __forceinline__ v8f wmma_bf16(v16bf a, v16bf b, v8f c) {
    return __builtin_amdgcn_wmma_f32_16x16x32_bf16(false, a, false, b,
                                                   (short)0, c, false, false);
}

__device__ __forceinline__ v8f v8f_zero() {
    v8f z;
#pragma unroll
    for (int v = 0; v < 8; ++v) z[v] = 0.0f;
    return z;
}

// ---- kernel 0: decay tables  powt[t][r] = gamma_r^t, powinv[t][r] = gamma_r^-t
__global__ void k_pow(const float* __restrict__ gamma, float* __restrict__ powt,
                      float* __restrict__ powinv) {
    int r = threadIdx.x;
    if (r < Rx) {
        float g = fmaxf(gamma[r], 1e-8f);
        float gi = 1.0f / g;
        float p = 1.0f, pi = 1.0f;
        for (int t = 0; t <= 128; ++t) {
            powt[t * Rx + r] = p;
            powinv[t * Rx + r] = pi;
            p *= g;
            pi *= gi;
        }
    }
}

// ---- kernel 1: per-block state increment  S[b,n,r,d] = sum_j g_r^(127-j) k[j,r] h[j,d]
//      also exports hTbf[b,n][d][j] = bf16(h[j,d])  (transposed, for k_output)
__global__ __launch_bounds__(256) void k_blockstate(const float* __restrict__ k,
                                                    const float* __restrict__ h,
                                                    const float* __restrict__ powt,
                                                    float* __restrict__ S,
                                                    __bf16* __restrict__ hTbf) {
    __shared__ __bf16 kT[128][132];  // [r][j], scaled by gamma^(127-j)
    __shared__ __bf16 hT[64][132];   // [d_local][j], one d-half at a time

    const int n = blockIdx.x, b = blockIdx.y;
    const int tid = threadIdx.x, lane = tid & 31, w = tid >> 5;

    const float* kblk = k + (size_t)(b * Wx + n * BLKx) * Rx;
    const float* hblk = h + (size_t)(b * Wx + n * BLKx) * Dx;
    float* Sblk = S + (size_t)(b * NBx + n) * (Rx * Dx);
    __bf16* hTblk = hTbf + (size_t)(b * NBx + n) * (Dx * BLKx);

    // pull the H tile toward the caches while we stage K
    __builtin_prefetch(hblk + (size_t)tid * 64, 0, 1);

    for (int idx = tid; idx < 128 * 128; idx += 256) {
        int r = idx & 127, j = idx >> 7;
        float val = kblk[idx] * powt[(127 - j) * Rx + r];
        kT[r][j] = (__bf16)val;
    }

    const int mBase = 16 * w;  // this wave's r-tile
    const int halfl = lane >> 4, nl = lane & 15;

    for (int halfd = 0; halfd < 2; ++halfd) {
        __syncthreads();
        for (int idx = tid; idx < 64 * 128; idx += 256) {
            int d = idx & 63, j = idx >> 6;
            hT[d][j] = (__bf16)hblk[(size_t)j * Dx + 64 * halfd + d];
        }
        __syncthreads();

        // export transposed bf16 H (coalesced along j)
        for (int idx = tid; idx < 64 * 128; idx += 256) {
            int j = idx & 127, d = idx >> 7;
            hTblk[(size_t)(64 * halfd + d) * BLKx + j] = hT[d][j];
        }

        v8f acc[4];
#pragma unroll
        for (int t = 0; t < 4; ++t) acc[t] = v8f_zero();

#pragma unroll
        for (int kt = 0; kt < 4; ++kt) {
            v16bf a = loadA_bf16(&kT[0][0], 132, mBase, 32 * kt, lane);
#pragma unroll
            for (int t = 0; t < 4; ++t) {
                v16bf bf = loadB_bf16(&hT[0][0], 132, 16 * t, 32 * kt, lane);
                acc[t] = wmma_bf16(a, bf, acc[t]);
            }
        }
#pragma unroll
        for (int t = 0; t < 4; ++t) {
            int dcol = 64 * halfd + 16 * t + nl;
#pragma unroll
            for (int v = 0; v < 8; ++v) {
                int rrow = mBase + v + 8 * halfl;
                Sblk[(size_t)rrow * Dx + dcol] = acc[t][v];
            }
        }
    }
}

// ---- kernel 2: in-place scan; S[b,n] becomes the state *entering* block n ---
__global__ __launch_bounds__(256) void k_scan(const float* __restrict__ powt,
                                              float* __restrict__ S) {
    int e = blockIdx.x * blockDim.x + threadIdx.x;
    if (e >= Bx * Rx * Dx) return;
    int d = e & 127, r = (e >> 7) & 127, b = e >> 14;
    float g128 = powt[128 * Rx + r];  // gamma_r^128
    float p = 0.0f;
    float* base = S + ((size_t)b * NBx) * (Rx * Dx) + (size_t)r * Dx + d;
    for (int nn = 0; nn < NBx; ++nn) {
        float* ptr = base + (size_t)nn * (Rx * Dx);
        float s = *ptr;
        *ptr = p;          // incoming state for block nn
        p = g128 * p + s;  // state after block nn
    }
}

// ---- kernel 3: outputs ------------------------------------------------------
// att[i,j] = sum_r (q[i,r] g^i)(k[j,r] g^-j), causally masked  (phase 1)
// out[i,d] = sum_j att[i,j] h[j,d] + sum_r q[i,r] g^(i+1) P[r,d] (phase 2)
__global__ __launch_bounds__(256) void k_output(const float* __restrict__ q,
                                                const float* __restrict__ k,
                                                const __bf16* __restrict__ hTbf,
                                                const float* __restrict__ powt,
                                                const float* __restrict__ powinv,
                                                const float* __restrict__ P,
                                                float* __restrict__ out) {
    __shared__ __bf16 att[128][132];  // attention matrix [i][j]
    __shared__ __bf16 buf[64][136];   // time-multiplexed: hT[d][j] then PT[d][r]
                                      // stride 136 -> 272B rows (16B aligned for B128)

    const int n = blockIdx.x, b = blockIdx.y;
    const int tid = threadIdx.x, lane = tid & 31, w = tid >> 5;
    const int halfl = lane >> 4, nl = lane & 15;

    const float* qblk = q + (size_t)(b * Wx + n * BLKx) * Rx;
    const float* kblk = k + (size_t)(b * Wx + n * BLKx) * Rx;
    const __bf16* hTblk = hTbf + (size_t)(b * NBx + n) * (Dx * BLKx);
    const float* Pblk = P + (size_t)(b * NBx + n) * (Rx * Dx);

    // pull H(bf16) and P tiles toward the caches while phase-1 attention runs
    __builtin_prefetch(hTblk + (size_t)tid * 64, 0, 1);
    __builtin_prefetch(Pblk + (size_t)tid * 64, 0, 1);

    const unsigned bufBase = (unsigned)(uintptr_t)&buf[0][0];

    // A fragment: qs[i,r] = q[i,r] * gamma_r^(i + rowOff)
    auto loadA_qs = [&](int mBase, int kBase, int rowOff) -> v16bf {
        int m = mBase + (lane & 15);
        const float* qrow = qblk + (size_t)m * Rx;
        const float* prow = powt + (size_t)(m + rowOff) * Rx;
        Frag f;
#pragma unroll
        for (int v = 0; v < 8; ++v) {
            int kk = kBase + 2 * v + 8 * halfl + (v >= 4 ? 8 : 0);
            f.u[v] = pack_bf16(qrow[kk] * prow[kk], qrow[kk + 1] * prow[kk + 1]);
        }
        return f.v;
    };
    // B fragment: ks[j,r] = k[j,r] * gamma_r^(-j)   (powinv table, no rcp)
    auto loadB_ks = [&](int nBase, int kBase) -> v16bf {
        int j = nBase + (lane & 15);
        const float* krow = kblk + (size_t)j * Rx;
        const float* prow = powinv + (size_t)j * Rx;
        Frag f;
#pragma unroll
        for (int v = 0; v < 8; ++v) {
            int kk = kBase + 2 * v + 8 * halfl + (v >= 4 ? 8 : 0);
            f.u[v] = pack_bf16(krow[kk] * prow[kk], krow[kk + 1] * prow[kk + 1]);
        }
        return f.v;
    };

    const int iT = w;

    // hoisted A fragments (q-hat, invariant across jT)
    v16bf aQ[4];
#pragma unroll
    for (int kt = 0; kt < 4; ++kt) aQ[kt] = loadA_qs(16 * iT, 32 * kt, 0);

    // ---- phase 1: build att in LDS (wave w owns i-tile w, all 8 j-tiles) ----
    for (int jT = 0; jT < 8; ++jT) {
        v8f acc = v8f_zero();
        if (jT <= iT) {
#pragma unroll
            for (int kt = 0; kt < 4; ++kt) {
                v16bf bf = loadB_ks(16 * jT, 32 * kt);
                acc = wmma_bf16(aQ[kt], bf, acc);
            }
        }
#pragma unroll
        for (int v = 0; v < 8; ++v) {
            int il = v + 8 * halfl;
            float val = (16 * iT + il >= 16 * jT + nl) ? acc[v] : 0.0f;
            att[16 * iT + il][16 * jT + nl] = (__bf16)val;
        }
    }

    // hoisted A fragments for the cross term (q * gamma^(i+1))
    v16bf aQG[4];
#pragma unroll
    for (int kt = 0; kt < 4; ++kt) aQG[kt] = loadA_qs(16 * iT, 32 * kt, 1);

    // ---- phase 2: per d-half: intra (att @ H) then cross ((q g^(i+1)) @ P) --
    float* oblk = out + (size_t)(b * Wx + n * BLKx) * Dx;

    for (int half = 0; half < 2; ++half) {
        __syncthreads();  // att ready (iter 0) / previous buf readers done

        // async-copy hT d-half (bf16, pre-transposed) into padded LDS rows
        for (int c = tid; c < 1024; c += 256) {
            int d = c >> 4, jc = (c & 15) * 8;  // 8 bf16 = 16B per chunk
            const __bf16* src = hTblk + (size_t)(64 * half + d) * BLKx + jc;
            unsigned ldsA = bufBase + (unsigned)(d * 136 + jc) * 2;
            asm volatile("global_load_async_to_lds_b128 %0, %1, off"
                         :: "v"(ldsA), "v"((unsigned long long)(uintptr_t)src)
                         : "memory");
        }
        asm volatile("s_wait_asynccnt 0x0" ::: "memory");
        __syncthreads();

        v8f acc2[4];
#pragma unroll
        for (int t = 0; t < 4; ++t) acc2[t] = v8f_zero();

        // intra-block: att @ H
#pragma unroll
        for (int kt = 0; kt < 4; ++kt) {
            v16bf a = loadA_bf16(&att[0][0], 132, 16 * iT, 32 * kt, lane);
#pragma unroll
            for (int t = 0; t < 4; ++t) {
                v16bf bf = loadB_bf16(&buf[0][0], 136, 16 * t, 32 * kt, lane);
                acc2[t] = wmma_bf16(a, bf, acc2[t]);
            }
        }

        __syncthreads();  // all waves done reading buf(h)
        // stage buf = PT: [d_local][r]  (f32 -> bf16 convert path)
        for (int idx = tid; idx < 64 * 128; idx += 256) {
            int d = idx & 63, r = idx >> 6;
            buf[d][r] = (__bf16)Pblk[(size_t)r * Dx + 64 * half + d];
        }
        __syncthreads();

        // cross-block: (q * gamma^(i+1)) @ P
#pragma unroll
        for (int kt = 0; kt < 4; ++kt) {
#pragma unroll
            for (int t = 0; t < 4; ++t) {
                v16bf bf = loadB_bf16(&buf[0][0], 136, 16 * t, 32 * kt, lane);
                acc2[t] = wmma_bf16(aQG[kt], bf, acc2[t]);
            }
        }

        // store this d-half
#pragma unroll
        for (int t = 0; t < 4; ++t) {
            int col = 64 * half + 16 * t + nl;
#pragma unroll
            for (int v = 0; v < 8; ++v) {
                int row = 16 * iT + v + 8 * halfl;
                oblk[(size_t)row * Dx + col] = acc2[t][v];
            }
        }
    }
}

extern "C" void kernel_launch(void* const* d_in, const int* in_sizes, int n_in,
                              void* d_out, int out_size, void* d_ws, size_t ws_size,
                              hipStream_t stream) {
    (void)in_sizes; (void)n_in; (void)out_size; (void)ws_size;
    const float* q     = (const float*)d_in[0];  // [B,W,R]
    const float* k     = (const float*)d_in[1];  // [B,W,R]
    const float* h     = (const float*)d_in[2];  // [B,W,D]
    const float* gamma = (const float*)d_in[3];  // [R]
    float* out = (float*)d_out;

    char* ws = (char*)d_ws;
    const size_t powBytes = (size_t)132 * Rx * sizeof(float);
    float*  powt   = (float*)ws;
    float*  powinv = (float*)(ws + powBytes);
    float*  S      = (float*)(ws + 2 * powBytes);                    // [B,NB,R,D] f32
    __bf16* hTbf   = (__bf16*)(ws + 2 * powBytes +
                               (size_t)Bx * NBx * Rx * Dx * sizeof(float)); // [B,NB,D,BLK]

    k_pow<<<1, 128, 0, stream>>>(gamma, powt, powinv);
    dim3 grid(NBx, Bx);
    k_blockstate<<<grid, 256, 0, stream>>>(k, h, powt, S, hTbf);
    k_scan<<<(Bx * Rx * Dx) / 256, 256, 0, stream>>>(powt, S);
    k_output<<<grid, 256, 0, stream>>>(q, k, hTbf, powt, powinv, S, out);
}
